// MultiHeadAttention_7980049236363
// MI455X (gfx1250) — matrix operand order repaired and checked by reading, not compiled, not run on hardware
//
#include <hip/hip_runtime.h>

// MHA: B=2, S=2048, E=1024, H=16, dh=64. fp32 in/out, bf16 WMMA internally.
// ws layout (bytes):
//   [0,8M)    Wt  : bf16 WqT,WkT,WvT,WoT  (each [j,e], 1024x1024)
//   [8M,16M)  xb  : bf16 x [4096,1024]    (reused as ctx after attention)
//   [16M,24M) Qb  : bf16 [B,H,S,dh]
//   [24M,32M) Kb  : bf16 [B,H,S,dh]
//   [32M,40M) Vtb : bf16 [B,H,dh,S]

#define E_DIM 1024
#define HEADS 16
#define DHEAD 64
#define BATCH 2
#define SEQ   2048

typedef __attribute__((ext_vector_type(16))) __bf16 v16bf;
typedef __attribute__((ext_vector_type(8)))  float  v8f;

union FragU { v16bf v; uint4 q[2]; };

static __device__ inline v16bf load_frag2(const __bf16* p0, const __bf16* p1) {
    FragU f;
    f.q[0] = *reinterpret_cast<const uint4*>(p0);
    f.q[1] = *reinterpret_cast<const uint4*>(p1);
    return f.v;
}

static __device__ inline v8f wmma_bf16(v16bf a, v16bf b, v8f c) {
    return __builtin_amdgcn_wmma_f32_16x16x32_bf16(false, a, false, b, (short)0, c,
                                                   false, false);
}

// ---------------- kernel 0a: weight convert + transpose (fp32 [e,j] -> bf16 [j,e])
__global__ void cvt_w_kernel(const float* __restrict__ W0, const float* __restrict__ W1,
                             const float* __restrict__ W2, const float* __restrict__ W3,
                             __bf16* __restrict__ Wt) {
    int w = blockIdx.y;
    const float* W = (w == 0) ? W0 : (w == 1) ? W1 : (w == 2) ? W2 : W3;
    int idx = blockIdx.x * blockDim.x + threadIdx.x;   // over 1024*1024
    int e = idx >> 10, j = idx & 1023;
    Wt[(size_t)w * E_DIM * E_DIM + (size_t)j * E_DIM + e] = (__bf16)W[idx];
}

// ---------------- kernel 0b: x fp32 -> bf16
__global__ void cvt_x_kernel(const float* __restrict__ x, __bf16* __restrict__ xb) {
    int idx = blockIdx.x * blockDim.x + threadIdx.x;
    xb[idx] = (__bf16)x[idx];
}

// ---------------- kernel 1: fused QKV projection.
// 128 threads = 4 waves, each wave one 64x64 tile of a 128x128 macrotile.
__global__ __launch_bounds__(128, 1) void qkv_gemm_kernel(
        const __bf16* __restrict__ xb, const __bf16* __restrict__ Wt,
        const float* __restrict__ bq, const float* __restrict__ bk,
        const float* __restrict__ bv,
        __bf16* __restrict__ Qb, __bf16* __restrict__ Kb, __bf16* __restrict__ Vtb) {
    int z = blockIdx.z;                                   // 0=Q 1=K 2=V
    const __bf16* W = Wt + (size_t)z * E_DIM * E_DIM;
    const float* bias = (z == 0) ? bq : (z == 1) ? bk : bv;
    int wid = threadIdx.x >> 5;
    int lane = threadIdx.x & 31;
    int ib = blockIdx.x * 128 + (wid & 1) * 64;
    int jb = blockIdx.y * 128 + (wid >> 1) * 64;
    int lrow = lane & 15, lhalf = lane >> 4;

    v8f acc[4][4];
#pragma unroll
    for (int mi = 0; mi < 4; mi++)
#pragma unroll
        for (int ni = 0; ni < 4; ni++) acc[mi][ni] = v8f{};

    for (int kc = 0; kc < E_DIM; kc += 32) {
        v16bf Bf[4];
#pragma unroll
        for (int ni = 0; ni < 4; ni++) {
            const __bf16* p = W + (size_t)(jb + ni * 16 + lrow) * E_DIM + kc + lhalf * 16;
            Bf[ni] = load_frag2(p, p + 8);
        }
#pragma unroll
        for (int mi = 0; mi < 4; mi++) {
            const __bf16* p = xb + (size_t)(ib + mi * 16 + lrow) * E_DIM + kc + lhalf * 16;
            v16bf A = load_frag2(p, p + 8);
#pragma unroll
            for (int ni = 0; ni < 4; ni++)
                acc[mi][ni] = wmma_bf16(A, Bf[ni], acc[mi][ni]);
        }
    }

#pragma unroll
    for (int ni = 0; ni < 4; ni++) {
        int j = jb + ni * 16 + lrow;
        float bb = bias[j];
        int h = j >> 6, d = j & 63;
#pragma unroll
        for (int mi = 0; mi < 4; mi++) {
#pragma unroll
            for (int r = 0; r < 8; r++) {
                int i = ib + mi * 16 + r + 8 * lhalf;
                int b = i >> 11, s = i & (SEQ - 1);
                __bf16 o = (__bf16)(acc[mi][ni][r] + bb);
                if (z == 0)
                    Qb[((size_t)(b * HEADS + h) * SEQ + s) * DHEAD + d] = o;
                else if (z == 1)
                    Kb[((size_t)(b * HEADS + h) * SEQ + s) * DHEAD + d] = o;
                else
                    Vtb[((size_t)(b * HEADS + h) * DHEAD + d) * SEQ + s] = o;
            }
        }
    }
}

// ---------------- kernel 2: flash attention. One wave per (b,h,16 q-rows).
__global__ __launch_bounds__(32, 1) void flash_attn_kernel(
        const __bf16* __restrict__ Qb, const __bf16* __restrict__ Kb,
        const __bf16* __restrict__ Vtb, __bf16* __restrict__ ctxb) {
    __shared__ __align__(16) __bf16 Plds[16 * 32];
    int bh = blockIdx.y;
    int qb = blockIdx.x * 16;
    int lane = threadIdx.x;
    int lrow = lane & 15, lhalf = lane >> 4;
    int b = bh >> 4, h = bh & 15;

    const __bf16* Qp = Qb + ((size_t)bh * SEQ + qb) * DHEAD;
    const __bf16* Kp = Kb + (size_t)bh * SEQ * DHEAD;
    const __bf16* Vp = Vtb + (size_t)bh * DHEAD * SEQ;

    v16bf Aq[2];
#pragma unroll
    for (int c = 0; c < 2; c++) {
        const __bf16* p = Qp + lrow * DHEAD + c * 32 + lhalf * 16;
        Aq[c] = load_frag2(p, p + 8);
    }

    v8f O[4];
#pragma unroll
    for (int t = 0; t < 4; t++) O[t] = v8f{};
    float mrun[8], lrun[8];
#pragma unroll
    for (int r = 0; r < 8; r++) { mrun[r] = -1e30f; lrun[r] = 0.f; }

    for (int kb = 0; kb < SEQ; kb += 32) {
        v8f s0 = v8f{}, s1 = v8f{};
#pragma unroll
        for (int c = 0; c < 2; c++) {
            const __bf16* p0 = Kp + (size_t)(kb + lrow) * DHEAD + c * 32 + lhalf * 16;
            s0 = wmma_bf16(Aq[c], load_frag2(p0, p0 + 8), s0);
            const __bf16* p1 = Kp + (size_t)(kb + 16 + lrow) * DHEAD + c * 32 + lhalf * 16;
            s1 = wmma_bf16(Aq[c], load_frag2(p1, p1 + 8), s1);
        }
        const float sc = 0.125f;  // 1/sqrt(64)
#pragma unroll
        for (int r = 0; r < 8; r++) { s0[r] *= sc; s1[r] *= sc; }

#pragma unroll
        for (int r = 0; r < 8; r++) {
            float cm = fmaxf(s0[r], s1[r]);
            cm = fmaxf(cm, __shfl_xor(cm, 1));
            cm = fmaxf(cm, __shfl_xor(cm, 2));
            cm = fmaxf(cm, __shfl_xor(cm, 4));
            cm = fmaxf(cm, __shfl_xor(cm, 8));
            float mnew = fmaxf(mrun[r], cm);
            float alpha = __expf(mrun[r] - mnew);
            mrun[r] = mnew;
            float p0v = __expf(s0[r] - mnew);
            float p1v = __expf(s1[r] - mnew);
            s0[r] = p0v; s1[r] = p1v;
            float rs = p0v + p1v;
            rs += __shfl_xor(rs, 1);
            rs += __shfl_xor(rs, 2);
            rs += __shfl_xor(rs, 4);
            rs += __shfl_xor(rs, 8);
            lrun[r] = lrun[r] * alpha + rs;
            O[0][r] *= alpha; O[1][r] *= alpha; O[2][r] *= alpha; O[3][r] *= alpha;
        }

        // D-layout -> A-layout reshuffle of P via LDS
#pragma unroll
        for (int r = 0; r < 8; r++) {
            int row = r + 8 * lhalf;
            Plds[row * 32 + lrow]      = (__bf16)s0[r];
            Plds[row * 32 + 16 + lrow] = (__bf16)s1[r];
        }
        __syncthreads();
        const __bf16* pl = &Plds[lrow * 32 + lhalf * 16];
        v16bf Ap = load_frag2(pl, pl + 8);
        __syncthreads();

#pragma unroll
        for (int t = 0; t < 4; t++) {
            const __bf16* pv = Vp + (size_t)(t * 16 + lrow) * SEQ + kb + lhalf * 16;
            O[t] = wmma_bf16(Ap, load_frag2(pv, pv + 8), O[t]);
        }
    }

#pragma unroll
    for (int r = 0; r < 8; r++) {
        float inv = 1.0f / lrun[r];
        int row = qb + r + 8 * lhalf;
        size_t base = ((size_t)b * SEQ + row) * E_DIM + h * DHEAD;
#pragma unroll
        for (int t = 0; t < 4; t++)
            ctxb[base + t * 16 + lrow] = (__bf16)(O[t][r] * inv);
    }
}

// ---------------- kernel 3: output projection, fp32 out
__global__ __launch_bounds__(128, 1) void out_gemm_kernel(
        const __bf16* __restrict__ ctxb, const __bf16* __restrict__ WoT,
        const float* __restrict__ bo, float* __restrict__ out) {
    int wid = threadIdx.x >> 5;
    int lane = threadIdx.x & 31;
    int ib = blockIdx.x * 128 + (wid & 1) * 64;
    int jb = blockIdx.y * 128 + (wid >> 1) * 64;
    int lrow = lane & 15, lhalf = lane >> 4;

    v8f acc[4][4];
#pragma unroll
    for (int mi = 0; mi < 4; mi++)
#pragma unroll
        for (int ni = 0; ni < 4; ni++) acc[mi][ni] = v8f{};

    for (int kc = 0; kc < E_DIM; kc += 32) {
        v16bf Bf[4];
#pragma unroll
        for (int ni = 0; ni < 4; ni++) {
            const __bf16* p = WoT + (size_t)(jb + ni * 16 + lrow) * E_DIM + kc + lhalf * 16;
            Bf[ni] = load_frag2(p, p + 8);
        }
#pragma unroll
        for (int mi = 0; mi < 4; mi++) {
            const __bf16* p = ctxb + (size_t)(ib + mi * 16 + lrow) * E_DIM + kc + lhalf * 16;
            v16bf A = load_frag2(p, p + 8);
#pragma unroll
            for (int ni = 0; ni < 4; ni++)
                acc[mi][ni] = wmma_bf16(A, Bf[ni], acc[mi][ni]);
        }
    }

#pragma unroll
    for (int ni = 0; ni < 4; ni++) {
        int j = jb + ni * 16 + lrow;
        float bb = bo[j];
#pragma unroll
        for (int mi = 0; mi < 4; mi++) {
#pragma unroll
            for (int r = 0; r < 8; r++) {
                int i = ib + mi * 16 + r + 8 * lhalf;
                out[(size_t)i * E_DIM + j] = acc[mi][ni][r] + bb;
            }
        }
    }
}

extern "C" void kernel_launch(void* const* d_in, const int* in_sizes, int n_in,
                              void* d_out, int out_size, void* d_ws, size_t ws_size,
                              hipStream_t stream) {
    (void)in_sizes; (void)n_in; (void)out_size; (void)ws_size;
    const float* x  = (const float*)d_in[0];
    const float* Wq = (const float*)d_in[1];
    const float* bq = (const float*)d_in[2];
    const float* Wk = (const float*)d_in[3];
    const float* bk = (const float*)d_in[4];
    const float* Wv = (const float*)d_in[5];
    const float* bv = (const float*)d_in[6];
    const float* Wo = (const float*)d_in[7];
    const float* bo = (const float*)d_in[8];
    float* out = (float*)d_out;

    char* ws = (char*)d_ws;
    const size_t MB = 1024ull * 1024ull;
    __bf16* Wt  = (__bf16*)(ws);            // 8 MB: WqT,WkT,WvT,WoT
    __bf16* xb  = (__bf16*)(ws + 8 * MB);   // 8 MB: x bf16, reused as ctx
    __bf16* Qb  = (__bf16*)(ws + 16 * MB);  // 8 MB
    __bf16* Kb  = (__bf16*)(ws + 24 * MB);  // 8 MB
    __bf16* Vtb = (__bf16*)(ws + 32 * MB);  // 8 MB

    cvt_w_kernel<<<dim3(4096, 4, 1), 256, 0, stream>>>(Wq, Wk, Wv, Wo, Wt);
    cvt_x_kernel<<<dim3(16384, 1, 1), 256, 0, stream>>>(x, xb);
    qkv_gemm_kernel<<<dim3(32, 8, 3), 128, 0, stream>>>(xb, Wt, bq, bk, bv, Qb, Kb, Vtb);
    flash_attn_kernel<<<dim3(128, 32, 1), 32, 0, stream>>>(Qb, Kb, Vtb, xb /*ctx*/);
    out_gemm_kernel<<<dim3(32, 8, 1), 128, 0, stream>>>(xb, Wt + 3ull * E_DIM * E_DIM, bo, out);
}
